// NodeLevelContrastiveLoss_6837587935550
// MI455X (gfx1250) — matrix operand order repaired and checked by reading, not compiled, not run on hardware
//
#include <hip/hip_runtime.h>

// Problem constants (from setup_inputs): G=256, NA=256, NO=512, D=128
#define G_CONST 256
#define NA_CONST 256
#define NO_CONST 512
#define D_CONST 128
#define TSTRIDE 68              // LDS row stride in dwords (68 mod 64 = 4 -> conflict-free B loads)
#define INV_TEMP 10.0f          // 1 / 0.1, folded into student normalization

typedef __attribute__((ext_vector_type(16))) __bf16 v16bf;
typedef __attribute__((ext_vector_type(8)))  float  v8f;

union FragBF {
    v16bf bf;
    unsigned int u[8];
    uint4 q[2];
};

__device__ __forceinline__ unsigned int pk2bf(float a, float b) {
    union { float f; unsigned int u; } x, y;
    x.f = a; y.f = b;
    unsigned int lo = (x.u + 0x7FFFu + ((x.u >> 16) & 1u)) >> 16;
    unsigned int hi = (y.u + 0x7FFFu + ((y.u >> 16) & 1u)) >> 16;
    return (hi << 16) | (lo & 0xFFFFu);
}

// Single-instruction butterfly step: ds_swizzle_b32 group-of-32 XOR pattern.
// offset = (xor_mask << 10) | or(0) | and(0x1F)
template <int XORMASK>
__device__ __forceinline__ float swz_add(float v) {
    int r = __builtin_amdgcn_ds_swizzle(__float_as_int(v), (XORMASK << 10) | 0x1F);
    return v + __int_as_float(r);
}

// One block per graph. 256 threads = 8 waves. Each wave owns TWO row-tiles of 16
// (A fragments for both resident in VGPRs) and sweeps 16 col-tiles; per jt:
// 8 ds_load_b128 feed 8 WMMAs (B shared by both row tiles), double-buffered,
// loop kept ROLLED to bound register pressure under 256 VGPRs.
// Softmax: scores bounded in [-10,10] (unit vectors / T) -> sum exp directly,
// lane-local partials, one ds_swizzle butterfly per row-tile at the end.
__global__ void __launch_bounds__(256, 1)
nlcl_main_kernel(const float* __restrict__ s_nodes,
                 const float* __restrict__ t_nodes,
                 const int*   __restrict__ kept,
                 float* __restrict__ partials) {
    __shared__ unsigned int t_tile[NA_CONST * TSTRIDE];  // gathered+normalized teacher rows, bf16 pairs
    __shared__ float s_inv[NA_CONST];                    // per-student-row 10/||x||

    const int g    = blockIdx.x;
    const int tid  = threadIdx.x;
    const int lane = tid & 31;
    const int wave = tid >> 5;
    const int half   = lane >> 4;
    const int lane16 = lane & 15;

    // ---- Stage teacher rows: gather via kept, L2-normalize, pack bf16 into LDS ----
    {
        const int r   = tid;                               // 0..255 (= NA rows)
        const int idx = kept[g * NA_CONST + r];            // local index into [0, NO)
        const float4* rp = (const float4*)(t_nodes + ((size_t)g * NO_CONST + (size_t)idx) * D_CONST);
        float ss = 0.0f;
        #pragma unroll 8
        for (int k = 0; k < D_CONST / 4; ++k) {
            float4 x = rp[k];
            ss = fmaf(x.x, x.x, ss); ss = fmaf(x.y, x.y, ss);
            ss = fmaf(x.z, x.z, ss); ss = fmaf(x.w, x.w, ss);
        }
        const float inv = 1.0f / fmaxf(sqrtf(ss), 1e-12f);
        unsigned int* dst = &t_tile[r * TSTRIDE];
        #pragma unroll 8
        for (int k = 0; k < D_CONST / 4; ++k) {
            float4 x = rp[k];
            dst[2 * k]     = pk2bf(x.x * inv, x.y * inv);
            dst[2 * k + 1] = pk2bf(x.z * inv, x.w * inv);
        }
    }
    // ---- Student row inverse norms (with 1/temperature folded in) ----
    {
        const float4* sp = (const float4*)(s_nodes + ((size_t)g * NA_CONST + tid) * D_CONST);
        float ss = 0.0f;
        #pragma unroll 8
        for (int k = 0; k < D_CONST / 4; ++k) {
            float4 x = sp[k];
            ss = fmaf(x.x, x.x, ss); ss = fmaf(x.y, x.y, ss);
            ss = fmaf(x.z, x.z, ss); ss = fmaf(x.w, x.w, ss);
        }
        s_inv[tid] = INV_TEMP / fmaxf(sqrtf(ss), 1e-12f);
    }
    __syncthreads();

    const int it0 = wave * 2;
    const int it1 = wave * 2 + 1;

    // Build A fragments for both row tiles (4 K-chunks each) from global fp32,
    // scaled + bf16-packed per the 16-bit A VGPR layout.
    FragBF afrag0[4], afrag1[4];
    #pragma unroll
    for (int rt = 0; rt < 2; ++rt) {
        FragBF* af = rt ? afrag1 : afrag0;
        const int arow = (it0 + rt) * 16 + lane16;
        const float ainv = s_inv[arow];
        const float4* arp4 = (const float4*)(s_nodes + ((size_t)g * NA_CONST + arow) * D_CONST);
        #pragma unroll
        for (int kk = 0; kk < 4; ++kk) {
            const int b0 = kk * 8 + half * 2;     // float4 index of K = kk*32 + half*8
            float4 x0 = arp4[b0];
            float4 x1 = arp4[b0 + 1];
            float4 x2 = arp4[b0 + 4];             // K + 16
            float4 x3 = arp4[b0 + 5];
            af[kk].u[0] = pk2bf(x0.x * ainv, x0.y * ainv);
            af[kk].u[1] = pk2bf(x0.z * ainv, x0.w * ainv);
            af[kk].u[2] = pk2bf(x1.x * ainv, x1.y * ainv);
            af[kk].u[3] = pk2bf(x1.z * ainv, x1.w * ainv);
            af[kk].u[4] = pk2bf(x2.x * ainv, x2.y * ainv);
            af[kk].u[5] = pk2bf(x2.z * ainv, x2.w * ainv);
            af[kk].u[6] = pk2bf(x3.x * ainv, x3.y * ainv);
            af[kk].u[7] = pk2bf(x3.z * ainv, x3.w * ainv);
        }
    }

    float sacc0[8], sacc1[8];
    #pragma unroll
    for (int r = 0; r < 8; ++r) { sacc0[r] = 0.0f; sacc1[r] = 0.0f; }

    // loss_acc collects -score[i,i] at diag lanes during the sweep,
    // and +log(sum) in the epilogue.
    float loss_acc = 0.0f;

    // B fragment loader: B layout (32x16 bf16): lane half h holds K=8h..8h+7
    // and K=16+8h..16+8h+7 -> two 4-dword groups per chunk.
    auto loadB = [&](FragBF* b, int jt) {
        const int j = jt * 16 + lane16;
        const unsigned int* base = &t_tile[j * TSTRIDE + half * 4];
        #pragma unroll
        for (int kk = 0; kk < 4; ++kk) {
            const uint4* p = (const uint4*)(base + kk * 16);
            b[kk].q[0] = p[0];
            b[kk].q[1] = p[2];
        }
    };

    auto compute = [&](FragBF* b, int jt) {
        v8f c0 = {}, c1 = {};
        #pragma unroll
        for (int kk = 0; kk < 4; ++kk) {
            c0 = __builtin_amdgcn_wmma_f32_16x16x32_bf16(
                    false, afrag0[kk].bf, false, b[kk].bf, (short)0, c0, false, false);
            c1 = __builtin_amdgcn_wmma_f32_16x16x32_bf16(
                    false, afrag1[kk].bf, false, b[kk].bf, (short)0, c1, false, false);
        }
        #pragma unroll
        for (int r = 0; r < 8; ++r) {
            sacc0[r] += __expf(c0[r]);
            sacc1[r] += __expf(c1[r]);
        }
        if (jt == it0) {                         // wave-uniform branch: diagonal tile
            #pragma unroll
            for (int r = 0; r < 8; ++r)
                if (lane == r + 24 * half) loss_acc -= c0[r];
        }
        if (jt == it1) {
            #pragma unroll
            for (int r = 0; r < 8; ++r)
                if (lane == r + 24 * half) loss_acc -= c1[r];
        }
    };

    // Double-buffered col-tile sweep; keep ROLLED (2 jt steps per iteration).
    FragBF b0[4], b1[4];
    loadB(b0, 0);
    #pragma clang loop unroll(disable)
    for (int jt = 0; jt < 16; jt += 2) {
        loadB(b1, jt + 1);
        compute(b0, jt);
        if (jt + 2 < 16) loadB(b0, jt + 2);
        compute(b1, jt + 1);
    }

    // One butterfly per row-tile: sum the 16 lane-partials of each half.
    // -logp[i,i] = log(sum_j exp(score[i,j])) - score[i,i]
    #pragma unroll
    for (int r = 0; r < 8; ++r) {
        float s0 = sacc0[r];
        float s1 = sacc1[r];
        s0 = swz_add<1>(s0); s0 = swz_add<2>(s0); s0 = swz_add<4>(s0); s0 = swz_add<8>(s0);
        s1 = swz_add<1>(s1); s1 = swz_add<2>(s1); s1 = swz_add<4>(s1); s1 = swz_add<8>(s1);
        if (lane == r + 24 * half)
            loss_acc += __logf(s0) + __logf(s1);
    }

    // Wave-level reduce, one partial per wave
    loss_acc = swz_add<1>(loss_acc);
    loss_acc = swz_add<2>(loss_acc);
    loss_acc = swz_add<4>(loss_acc);
    loss_acc = swz_add<8>(loss_acc);
    loss_acc = swz_add<16>(loss_acc);
    if (lane == 0)
        partials[g * 8 + wave] = loss_acc;
}

__global__ void nlcl_reduce_kernel(const float* __restrict__ partials, float* __restrict__ out) {
    __shared__ float wsum[8];
    const int tid  = threadIdx.x;
    const int lane = tid & 31;
    const int wave = tid >> 5;
    float a = 0.0f;
    for (int i = tid; i < G_CONST * 8; i += 256)
        a += partials[i];
    a = swz_add<1>(a);
    a = swz_add<2>(a);
    a = swz_add<4>(a);
    a = swz_add<8>(a);
    a = swz_add<16>(a);
    if (lane == 0) wsum[wave] = a;
    __syncthreads();
    if (tid == 0) {
        float t = 0.0f;
        #pragma unroll
        for (int i = 0; i < 8; ++i) t += wsum[i];
        out[0] = t * (1.0f / ((float)G_CONST * (float)NA_CONST));
    }
}

extern "C" void kernel_launch(void* const* d_in, const int* in_sizes, int n_in,
                              void* d_out, int out_size, void* d_ws, size_t ws_size,
                              hipStream_t stream) {
    const float* s_nodes = (const float*)d_in[0];
    const float* t_nodes = (const float*)d_in[1];
    const int*   kept    = (const int*)d_in[2];
    float* partials = (float*)d_ws;               // G*8 floats
    float* out      = (float*)d_out;

    hipLaunchKernelGGL(nlcl_main_kernel, dim3(G_CONST), dim3(256), 0, stream,
                       s_nodes, t_nodes, kept, partials);
    hipLaunchKernelGGL(nlcl_reduce_kernel, dim3(1), dim3(256), 0, stream,
                       partials, out);
}
